// HeadAttn_46050639347824
// MI455X (gfx1250) — compile-verified, matrix operand order checked by reading, not run on hardware
//
#include <hip/hip_runtime.h>
#include <hip/hip_bf16.h>
#include <stdint.h>

// ---------------- problem constants ----------------
#define HEADS 8
#define SEQ   2048
#define BATCH 4
#define DIMF  520          // padded feature dim (512 + 4 + 4)
#define DH    65           // head dim = 520/8
#define DHP   96           // DH padded to 3 K-chunks of 32 for bf16 WMMA
#define KIN   512          // projection reduction dim (padded channels are 0)
#define NPAD  528          // DIMF padded to 33 column tiles of 16
#define NJT   33

// LDS layout for the attention kernel (bytes)
#define SBUF_BYTES   (16 * SEQ * 4)              // 131072 f32 scores
#define PBUF_BYTES   (16 * SEQ * 2)              //  65536 bf16 probs
#define STAT_BYTES   (2 * 16 * 4)                //    128 rowmax/rowinv
#define KSTAGE_OFF   (SBUF_BYTES + PBUF_BYTES + STAT_BYTES)   // 196736
#define KTILE_BYTES  (16 * DHP * 2)              //   3072 one 16x96 bf16 tile
#define KSTAGE_BYTES (8 * 2 * KTILE_BYTES)       //  49152 double buffer / wave
#define SMEM_TOTAL   (KSTAGE_OFF + KSTAGE_BYTES) // 245888 (< 320KB WGP LDS)

typedef __attribute__((ext_vector_type(16))) __bf16          v16bf;
typedef __attribute__((ext_vector_type(16))) unsigned short  v16u;
typedef __attribute__((ext_vector_type(8)))  float           v8f;
typedef unsigned short u16;
typedef unsigned int   u32;

union BFV { v16u u; v16bf b; };

__device__ __forceinline__ u16 f2bf(float f) {
  u32 u = __float_as_uint(f);
  u += 0x7FFFu + ((u >> 16) & 1u);     // round-to-nearest-even
  return (u16)(u >> 16);
}

// Load one lane's 16 bf16 elements of a 16x32 A-tile / 32x16 B-tile K-chunk.
// Per CDNA5 ISA 16-bit layout, element i maps to
//   k = ((i>>3)<<4) + half*8 + ((i>>1)&3)*2 + (i&1)
// `base` is this lane's row-major source row at the chunk start, so each
// 8-element group is 16 contiguous bytes -> compiler merges to b128 loads.
__device__ __forceinline__ v16bf load_row16(const u16* base, int half) {
  BFV r;
#pragma unroll
  for (int i2 = 0; i2 < 8; ++i2) {
    int k0 = (((i2 >> 2) << 4)) + half * 8 + ((i2 & 3) << 1);
    u32 pair = *(const u32*)(base + k0);
    r.u[2 * i2]     = (u16)(pair & 0xFFFFu);
    r.u[2 * i2 + 1] = (u16)(pair >> 16);
  }
  return r.b;
}

__device__ __forceinline__ v8f wmma_bf16(v16bf a, v16bf b, v8f c) {
  return __builtin_amdgcn_wmma_f32_16x16x32_bf16(
      /*neg_a=*/false, a, /*neg_b=*/false, b,
      /*c_mod=*/(short)0, c, /*reuse_a=*/false, /*reuse_b=*/false);
}

// Async-copy one contiguous 3072B K tile (16 rows x 96 bf16) global -> LDS.
// 6 issues x (32 lanes x 16B); tracked on ASYNCcnt, consumed after
// s_wait_asynccnt. lds_off / gsrc already include this lane's +16B slot.
__device__ __forceinline__ void async_copy_ktile(unsigned lds_off, const u16* gsrc) {
  unsigned long long ga = (unsigned long long)(uintptr_t)gsrc;
#pragma unroll
  for (int i = 0; i < 6; ++i)
    asm volatile("global_load_async_to_lds_b128 %0, %1, off offset:%2"
                 :: "v"(lds_off), "v"(ga), "i"(i * 512) : "memory");
}

// ---------------- prep kernels ----------------
__global__ void zero_ws(u16* p, long n) {
  long i = (long)blockIdx.x * blockDim.x + threadIdx.x;
  long s = (long)gridDim.x * blockDim.x;
  for (; i < n; i += s) p[i] = 0;
}

__global__ void cvt_x(const float* __restrict__ x, u16* __restrict__ xb, int n) {
  int i = blockIdx.x * blockDim.x + threadIdx.x;
  if (i < n) xb[i] = f2bf(x[i]);
}

// w (520x520 row-major) -> wt (528 x 512) bf16, wt[j][d] = w[j][d+4]
__global__ void cvt_w(const float* __restrict__ w, u16* __restrict__ wt) {
  int i = blockIdx.x * blockDim.x + threadIdx.x;
  if (i >= NPAD * KIN) return;
  int j = i / KIN, d = i - j * KIN;
  wt[i] = (j < DIMF) ? f2bf(w[j * DIMF + d + 4]) : (u16)0;
}

// ---------------- QKV projection: one wave per 16x16 tile ----------------
__global__ void proj_qkv(const u16* __restrict__ xb, const u16* __restrict__ wt,
                         u16* __restrict__ dst, int transposed) {
  int nt = blockIdx.x;            // 0..32  (column tile over 528)
  int mt = blockIdx.y;            // 0..511 (row tile over B*SEQ)
  int lane = threadIdx.x & 31;
  int half = lane >> 4, r15 = lane & 15;

  const u16* arow = xb + (size_t)(mt * 16 + r15) * KIN;
  const u16* brow = wt + (size_t)(nt * 16 + r15) * KIN;

  v8f acc = {};
#pragma unroll 4
  for (int kc = 0; kc < KIN; kc += 32) {
    v16bf a = load_row16(arow + kc, half);
    v16bf b = load_row16(brow + kc, half);
    acc = wmma_bf16(a, b, acc);
  }
#pragma unroll
  for (int r = 0; r < 8; ++r) {
    int m  = half * 8 + r;
    int gr = mt * 16 + m;
    int bb = gr >> 11, l = gr & (SEQ - 1);
    int j  = nt * 16 + r15;
    if (j < DIMF) {
      int h = j / DH, dh = j - h * DH;
      u16 v = f2bf(acc[r]);
      if (!transposed)   // Q,K: (b,h,l,DHP)
        dst[((size_t)(bb * HEADS + h) * SEQ + l) * DHP + dh] = v;
      else               // V: (b,h,dh,SEQ) so PV B-tiles are row-contiguous
        dst[((size_t)(bb * HEADS + h) * DHP + dh) * SEQ + l] = v;
    }
  }
}

// ---------------- fused attention: one WG = one (b,h,16-query) tile --------
__global__ void __launch_bounds__(256) attn_kernel(
    const u16* __restrict__ qws, const u16* __restrict__ kws,
    const u16* __restrict__ vwsT, float* __restrict__ out,
    float* __restrict__ attn) {
  extern __shared__ char smem[];
  float* sbuf = (float*)smem;                       // 16 x 2048 f32 scores
  u16*   pbuf = (u16*)(smem + SBUF_BYTES);          // 16 x 2048 bf16 probs
  float* rmax = (float*)(smem + SBUF_BYTES + PBUF_BYTES);
  float* rinv = rmax + 16;

  int qt = blockIdx.x, h = blockIdx.y, b = blockIdx.z;
  int w = threadIdx.x >> 5, lane = threadIdx.x & 31;
  int half = lane >> 4, r15 = lane & 15;
  size_t bh = (size_t)(b * HEADS + h);

  // Q A-tiles resident in VGPRs: 3 K-chunks covering DHP=96
  const u16* qrow = qws + (bh * SEQ + qt * 16 + r15) * DHP;
  v16bf aq0 = load_row16(qrow + 0,  half);
  v16bf aq1 = load_row16(qrow + 32, half);
  v16bf aq2 = load_row16(qrow + 64, half);

  const float scale = 0.12403473458920847f;         // 1/sqrt(65)

  // ---- S = Q K^T with ASYNCcnt-tracked double-buffered K staging ----
  // This wave's two LDS tile buffers (hardware wants the LDS byte address;
  // take the low 32 bits of the generic pointer per the aperture rules).
  unsigned kbase = (unsigned)(uintptr_t)(smem + KSTAGE_OFF) +
                   (unsigned)w * 2u * KTILE_BYTES;
  const u16* ktile0 = kws + (bh * SEQ) * DHP;       // row 0 of K for this (b,h)

  // prefetch first tile (nt = w) into buffer 0
  async_copy_ktile(kbase + lane * 16, ktile0 + (size_t)w * 16 * DHP + lane * 8);

  int cur = 0;
  for (int it = 0; it < 16; ++it) {
    int nt  = w + it * 8;
    int nxt = nt + 8;
    if (nxt < SEQ / 16) {   // prefetch next tile into the other buffer
      async_copy_ktile(kbase + (cur ^ 1) * KTILE_BYTES + lane * 16,
                       ktile0 + (size_t)nxt * 16 * DHP + lane * 8);
      asm volatile("s_wait_asynccnt 0x6" ::: "memory");  // current tile landed
    } else {
      asm volatile("s_wait_asynccnt 0x0" ::: "memory");  // drain
    }
    const u16* kb = (const u16*)(smem + (kbase - (unsigned)(uintptr_t)smem) +
                                 cur * KTILE_BYTES);
    const u16* krow = kb + r15 * DHP;
    v8f acc = {};
    acc = wmma_bf16(aq0, load_row16(krow + 0,  half), acc);
    acc = wmma_bf16(aq1, load_row16(krow + 32, half), acc);
    acc = wmma_bf16(aq2, load_row16(krow + 64, half), acc);
#pragma unroll
    for (int r = 0; r < 8; ++r)
      sbuf[(half * 8 + r) * SEQ + nt * 16 + r15] = acc[r] * scale;
    cur ^= 1;
  }
  __syncthreads();

  // ---- exact softmax stats: each wave owns 2 rows (wave32 shuffles)
#pragma unroll
  for (int r2 = 0; r2 < 2; ++r2) {
    int row = w * 2 + r2;
    const float* srow = sbuf + row * SEQ;
    float mx = -3.4e38f;
    for (int c = lane; c < SEQ; c += 32) mx = fmaxf(mx, srow[c]);
#pragma unroll
    for (int off = 16; off >= 1; off >>= 1) mx = fmaxf(mx, __shfl_xor(mx, off));
    float sm = 0.f;
    for (int c = lane; c < SEQ; c += 32) sm += __expf(srow[c] - mx);
#pragma unroll
    for (int off = 16; off >= 1; off >>= 1) sm += __shfl_xor(sm, off);
    if (lane == 0) { rmax[row] = mx; rinv[row] = 1.0f / sm; }
  }
  __syncthreads();

  // ---- normalize: NT-stream attn to HBM (537MB write-once; keep L2 for K/V)
  float* aout = attn + (bh * SEQ + (size_t)qt * 16) * SEQ;
  for (int i = threadIdx.x; i < 16 * SEQ; i += 256) {
    int row = i >> 11, col = i & (SEQ - 1);
    float p = __expf(sbuf[i] - rmax[row]) * rinv[row];
    __builtin_nontemporal_store(p, &aout[(size_t)row * SEQ + col]);
    pbuf[i] = f2bf(p);
  }
  __syncthreads();

  // ---- O = P V: 6 waves cover DHP=96 (only dh<65 is real)
  if (w < 6) {
    const u16* vrow = vwsT + (bh * DHP + w * 16 + r15) * SEQ;
    v8f acc = {};
    for (int kc = 0; kc < SEQ; kc += 32) {
      v16bf a  = load_row16(pbuf + r15 * SEQ + kc, half);
      v16bf bm = load_row16(vrow + kc, half);
      acc = wmma_bf16(a, bm, acc);
    }
#pragma unroll
    for (int r = 0; r < 8; ++r) {
      int m  = half * 8 + r;
      int dh = w * 16 + r15;
      int j  = h * DH + dh;
      if (dh < DH && j >= 4 && j < 516)   // channel slice [4,516)
        __builtin_nontemporal_store(
            acc[r], &out[((size_t)b * SEQ + qt * 16 + m) * 512 + (j - 4)]);
    }
  }
}

// ---------------- host launcher ----------------
extern "C" void kernel_launch(void* const* d_in, const int* in_sizes, int n_in,
                              void* d_out, int out_size, void* d_ws, size_t ws_size,
                              hipStream_t stream) {
  (void)in_sizes; (void)n_in; (void)out_size; (void)ws_size;
  const float* inp = (const float*)d_in[0];
  const float* w_q = (const float*)d_in[1];
  const float* w_k = (const float*)d_in[2];
  const float* w_v = (const float*)d_in[3];
  float* out  = (float*)d_out;
  float* attn = out + (size_t)BATCH * SEQ * 512;

  char* ws = (char*)d_ws;
  size_t off = 0;
  auto take = [&](size_t bytes) -> char* {
    char* p = ws + off;
    off = (off + bytes + 255) & ~(size_t)255;
    return p;
  };
  u16* xb  = (u16*)take((size_t)BATCH * SEQ * KIN * 2);   // bf16 input
  u16* wtq = (u16*)take((size_t)NPAD * KIN * 2);
  u16* wtk = (u16*)take((size_t)NPAD * KIN * 2);
  u16* wtv = (u16*)take((size_t)NPAD * KIN * 2);
  u16* qws = (u16*)take((size_t)BATCH * HEADS * SEQ * DHP * 2);
  u16* kws = (u16*)take((size_t)BATCH * HEADS * SEQ * DHP * 2);
  u16* vws = (u16*)take((size_t)BATCH * HEADS * SEQ * DHP * 2); // transposed

  // zero Q/K/V (contiguous) so DHP padding contributes 0 to the GEMMs
  long zn = (long)3 * BATCH * HEADS * SEQ * DHP;
  zero_ws<<<1024, 256, 0, stream>>>(qws, zn);

  int nx = BATCH * SEQ * KIN;
  cvt_x<<<(nx + 255) / 256, 256, 0, stream>>>(inp, xb, nx);
  int nw = NPAD * KIN;
  cvt_w<<<(nw + 255) / 256, 256, 0, stream>>>(w_q, wtq);
  cvt_w<<<(nw + 255) / 256, 256, 0, stream>>>(w_k, wtk);
  cvt_w<<<(nw + 255) / 256, 256, 0, stream>>>(w_v, wtv);

  dim3 pg(NJT, (BATCH * SEQ) / 16);
  proj_qkv<<<pg, 32, 0, stream>>>(xb, wtq, qws, 0);
  proj_qkv<<<pg, 32, 0, stream>>>(xb, wtk, kws, 0);
  proj_qkv<<<pg, 32, 0, stream>>>(xb, wtv, vws, 1);

  dim3 ag(SEQ / 16, HEADS, BATCH);
  attn_kernel<<<ag, 256, SMEM_TOTAL, stream>>>(qws, kws, vws, out, attn);
}
